// VolumeRenderer_60232621359587
// MI455X (gfx1250) — compile-verified, matrix operand order checked by reading, not run on hardware
//
#include <hip/hip_runtime.h>
#include <hip/hip_bf16.h>

#define N_RAYS   262144
#define N_PTS    128
#define FAR_DELTA 1e10f
#define X_CLAMP   60.0f   // exp(-60) == 0 in f32; keeps f16 A-operand finite

typedef __attribute__((ext_vector_type(16))) _Float16 v16h;
typedef __attribute__((ext_vector_type(8)))  float    v8f;
typedef __attribute__((ext_vector_type(4)))  float    f4;

// One wave processes a tile of 16 rays x 128 points.
// Prefix sum S of x = delta*density is computed as X(16x32) * Ltri(32x16) + carry
// with v_wmma_f32_16x16x32_f16, then weights w_p = exp(-S_{p-1}) - exp(-S_p).
__global__ __launch_bounds__(256) void volrend_main(
    const float* __restrict__ depth_values,   // [N, 128]
    const float* __restrict__ density,        // [N, 128, 1]
    const float* __restrict__ feature,        // [N, 128, 3]
    float* __restrict__ feat_out,             // [N, 3]
    float* __restrict__ depth_out,            // [N] (raw, normalized by 2nd kernel)
    unsigned int* __restrict__ gmax)
{
    const int lane = threadIdx.x & 31;
    const int wave = threadIdx.x >> 5;
    const int half = lane >> 4;    // K-half for A/B operands
    const int col  = lane & 15;    // A: row M; B/D: column N
    const int tile = blockIdx.x * 8 + wave;
    const int rayBase = tile * 16;

    // Triangular B matrices (per-lane constants).
    // 16-bit operand slot map: vgpr j<4 -> K = 8*half + 2j + e ; j>=4 -> K = 16 + 8*half + 2(j-4) + e
    v16h b1, b2;
    #pragma unroll
    for (int j = 0; j < 8; ++j) {
        #pragma unroll
        for (int e = 0; e < 2; ++e) {
            const int K = (j < 4) ? (8 * half + 2 * j + e)
                                  : (16 + 8 * half + 2 * (j - 4) + e);
            b1[2 * j + e] = (_Float16)((K <= col)      ? 1.0f : 0.0f); // prefixes 0..15
            b2[2 * j + e] = (_Float16)((K <= col + 16) ? 1.0f : 0.0f); // prefixes 16..31
        }
    }

    // A-operand rows: this lane owns ray (rayBase + col) for its K slots.
    const float* rowd  = depth_values + (size_t)(rayBase + col) * N_PTS;
    const float* rowden = density     + (size_t)(rayBase + col) * N_PTS;

    v8f carry = {};            // running cumsum per accumulator row (8 rows / lane-half)
    float facc[3][8];
    #pragma unroll
    for (int ch = 0; ch < 3; ++ch)
        #pragma unroll
        for (int i = 0; i < 8; ++i) facc[ch][i] = 0.0f;

    int firstp = 1000;         // first p<50 with density > 0.1 (for this lane's slots)

    #pragma unroll
    for (int c = 0; c < 4; ++c) {
        const int pbase = c * 32;

        // ---- build A (f16) from two contiguous 8-point runs per lane ----
        v16h a;
        #pragma unroll
        for (int r = 0; r < 2; ++r) {
            const int p0 = pbase + (r ? (16 + 8 * half) : (8 * half));
            f4 q0 = *(const f4*)(rowd + p0);
            f4 q1 = *(const f4*)(rowd + p0 + 4);
            f4 s0 = *(const f4*)(rowden + p0);
            f4 s1 = *(const f4*)(rowden + p0 + 4);
            float d[9];
            d[0] = q0.x; d[1] = q0.y; d[2] = q0.z; d[3] = q0.w;
            d[4] = q1.x; d[5] = q1.y; d[6] = q1.z; d[7] = q1.w;
            const bool tail = (p0 + 8) >= N_PTS;       // only true for the very last run
            d[8] = tail ? 0.0f : rowd[p0 + 8];
            float den[8] = { s0.x, s0.y, s0.z, s0.w, s1.x, s1.y, s1.z, s1.w };
            #pragma unroll
            for (int i = 0; i < 8; ++i) {
                const float delta = (i == 7 && tail) ? FAR_DELTA : (d[i + 1] - d[i]);
                const float x = fminf(delta * den[i], X_CLAMP);
                a[r * 8 + i] = (_Float16)x;
                const int p = p0 + i;
                if (p < 50 && den[i] > 0.1f) firstp = min(firstp, p);
            }
        }

        // ---- prefix sums via WMMA: S = A x Ltri + carry ----
        v8f S1 = __builtin_amdgcn_wmma_f32_16x16x32_f16(
            false, a, false, b1, (short)0, carry, false, false);   // points pbase + 0..15
        v8f S2 = __builtin_amdgcn_wmma_f32_16x16x32_f16(
            false, a, false, b2, (short)0, carry, false, false);   // points pbase + 16..31

        // ---- weights + feature accumulation (D-layout: row = i + 8*half, col = lane%16) ----
        #pragma unroll
        for (int i = 0; i < 8; ++i) {
            const float ecarry = __expf(-carry[i]);
            const float e1 = __expf(-S1[i]);
            const float e2 = __expf(-S2[i]);
            float e1prev = __shfl_up(e1, 1, 16);
            float e2prev = __shfl_up(e2, 1, 16);
            const float e1last = __shfl(e1, 15, 16);
            if (col == 0) { e1prev = ecarry; e2prev = e1last; }
            const float w1 = e1prev - e1;
            const float w2 = e2prev - e2;
            carry[i] = __shfl(S2[i], 15, 16);          // S at point pbase+31, broadcast

            const int m = rayBase + i + 8 * half;
            const float* f1 = feature + ((size_t)m * N_PTS + (size_t)(pbase + col)) * 3;
            const float* f2 = f1 + 16 * 3;
            facc[0][i] += w1 * f1[0] + w2 * f2[0];
            facc[1][i] += w1 * f1[1] + w2 * f2[1];
            facc[2][i] += w1 * f1[2] + w2 * f2[2];
        }
    }

    // ---- depth: combine the two lane-halves covering the same ray ----
    const int fo = min(firstp, __shfl_xor(firstp, 16));
    const float draw = (fo < 50) ? (1.0f - (float)fo * 0.02f) : 0.0f;
    if (half == 0) depth_out[rayBase + col] = draw;
    float dmax = draw;
    #pragma unroll
    for (int o = 8; o > 0; o >>= 1) dmax = fmaxf(dmax, __shfl_xor(dmax, o, 16));
    if (lane == 0) atomicMax(gmax, __float_as_uint(dmax));   // values >= 0: uint order == float order

    // ---- feature: reduce across the 16 columns of each accumulator row ----
    #pragma unroll
    for (int ch = 0; ch < 3; ++ch) {
        #pragma unroll
        for (int i = 0; i < 8; ++i) {
            float v = facc[ch][i];
            #pragma unroll
            for (int o = 8; o > 0; o >>= 1) v += __shfl_xor(v, o, 16);
            if (col == 0)
                feat_out[(size_t)(rayBase + i + 8 * half) * 3 + ch] = v;
        }
    }
}

__global__ void volrend_init(unsigned int* gmax) { *gmax = 0u; }

__global__ void volrend_norm(float* __restrict__ depth,
                             const unsigned int* __restrict__ gmax, int n)
{
    const int i = blockIdx.x * blockDim.x + threadIdx.x;
    const float m = __uint_as_float(*gmax);
    if (i < n) depth[i] = depth[i] / m;
}

extern "C" void kernel_launch(void* const* d_in, const int* in_sizes, int n_in,
                              void* d_out, int out_size, void* d_ws, size_t ws_size,
                              hipStream_t stream) {
    const float* depth_values = (const float*)d_in[0];   // [N, 128]
    const float* density      = (const float*)d_in[1];   // [N, 128, 1]
    const float* feature      = (const float*)d_in[2];   // [N, 128, 3]

    float* feat_out  = (float*)d_out;                    // first N*3 floats
    float* depth_out = feat_out + (size_t)N_RAYS * 3;    // next N floats
    unsigned int* gmax = (unsigned int*)d_ws;

    hipLaunchKernelGGL(volrend_init, dim3(1), dim3(1), 0, stream, gmax);

    // 16 rays/wave, 8 waves/block -> 128 rays/block
    hipLaunchKernelGGL(volrend_main, dim3(N_RAYS / 128), dim3(256), 0, stream,
                       depth_values, density, feature, feat_out, depth_out, gmax);

    hipLaunchKernelGGL(volrend_norm, dim3((N_RAYS + 255) / 256), dim3(256), 0, stream,
                       depth_out, gmax, N_RAYS);
}
